// MultiViewAttentionLSTMBasedRegressor_55353538511277
// MI455X (gfx1250) — compile-verified, matrix operand order checked by reading, not run on hardware
//
#include <hip/hip_runtime.h>
#include <hip/hip_bf16.h>

#define G_GENES 2048
#define T_PTS   512
#define H_DIM   64
#define L4_DIM  256

typedef __attribute__((ext_vector_type(16))) __bf16 v16bf;
typedef __attribute__((ext_vector_type(8)))  float  v8f;

// ---- transcendentals: use native v_tanh_f32 on gfx1250 when available ----
#if __has_builtin(__builtin_amdgcn_tanhf)
__device__ __forceinline__ float fast_tanh(float x) {
    return __builtin_amdgcn_tanhf(x);
}
#else
__device__ __forceinline__ float fast_tanh(float x) {
    return 1.0f - 2.0f * __builtin_amdgcn_rcpf(1.0f + __expf(2.0f * x));
}
#endif
__device__ __forceinline__ float fast_sig(float x) {
    // sigmoid(x) = 0.5*tanh(x/2) + 0.5  (single trans op with v_tanh_f32)
    return fmaf(fast_tanh(0.5f * x), 0.5f, 0.5f);
}
__device__ __forceinline__ unsigned short f2bf(float f) {
    unsigned u = __float_as_uint(f);
    u = (u + 0x7FFFu + ((u >> 16) & 1u)) >> 16;  // round-to-nearest-even
    return (unsigned short)u;
}
__device__ __forceinline__ unsigned pack_bf16(float lo, float hi) {
    return (unsigned)f2bf(lo) | ((unsigned)f2bf(hi) << 16);
}
template<int PAT>
__device__ __forceinline__ float swz_add(float v) {
    return v + __int_as_float(__builtin_amdgcn_ds_swizzle(__float_as_int(v), PAT));
}

// ---- CDNA5 async memory->LDS (GLOBAL_LOAD_ASYNC_TO_LDS_B32, ASYNCcnt) ----
__device__ __forceinline__ void async_ld_b32(unsigned lds_off, const void* gaddr) {
    asm volatile("global_load_async_to_lds_b32 %0, %1, off"
                 :: "v"(lds_off), "v"((unsigned long long)(uintptr_t)gaddr)
                 : "memory");
}
__device__ __forceinline__ void wait_async0() {
    asm volatile("s_wait_asynccnt 0" ::: "memory");
}
__device__ __forceinline__ void wait_ds0() {
    asm volatile("s_wait_dscnt 0" ::: "memory");
}
__device__ __forceinline__ unsigned lds_off(const void* p) {
    return (unsigned)(uintptr_t)p;   // generic LDS addr: low 32 bits = LDS offset
}

// ---------------------------------------------------------------------------
// Kernel 1: unmasked pass -> h_all[T,64]. 0.5 GFLOP, one block per timepoint.
// ---------------------------------------------------------------------------
__global__ __launch_bounds__(256)
void k1_hall(const float* __restrict__ x,
             const float* __restrict__ Wih0,
             const float* __restrict__ bih0, const float* __restrict__ bhh0,
             const float* __restrict__ Wih1,
             const float* __restrict__ bih1, const float* __restrict__ bhh1,
             float* __restrict__ h_all) {
    __shared__ float xcol[G_GENES];
    __shared__ float gates[L4_DIM];
    __shared__ float h0v[H_DIM];
    const int t = blockIdx.x, tid = threadIdx.x;
    for (int g = tid; g < G_GENES; g += 256) xcol[g] = x[g * T_PTS + t];
    __syncthreads();
    {
        float acc = bih0[tid] + bhh0[tid];
        const float* w = Wih0 + tid * G_GENES;
        #pragma unroll 8
        for (int g = 0; g < G_GENES; ++g) acc += w[g] * xcol[g];
        gates[tid] = acc;
    }
    __syncthreads();
    if (tid < H_DIM) {
        float cc = fast_sig(gates[tid]) * fast_tanh(gates[128 + tid]);
        h0v[tid] = fast_sig(gates[192 + tid]) * fast_tanh(cc);
    }
    __syncthreads();
    {
        float acc = bih1[tid] + bhh1[tid];
        const float* w = Wih1 + tid * H_DIM;
        #pragma unroll
        for (int h = 0; h < H_DIM; ++h) acc += w[h] * h0v[h];
        __syncthreads();             // gates[] reads above must finish
        gates[tid] = acc;
    }
    __syncthreads();
    if (tid < H_DIM) {
        float cc = fast_sig(gates[tid]) * fast_tanh(gates[128 + tid]);
        h_all[t * H_DIM + tid] = fast_sig(gates[192 + tid]) * fast_tanh(cc);
    }
}

// ---------------------------------------------------------------------------
// Main WMMA kernel. PASS 0: h tiles -> attention logits[T,G].
//                   PASS 1: recompute h tiles, accumulate aw-weighted h_att.
// grid = (T/16, G/64), block = 256 (8 wave32s, each wave walks 8 genes).
// ---------------------------------------------------------------------------
template<int PASS>
__global__ __launch_bounds__(256, 2)
void k_main(const float* __restrict__ x,
            const float* __restrict__ Wih0,
            const float* __restrict__ bih0, const float* __restrict__ bhh0,
            const float* __restrict__ Wih1,
            const float* __restrict__ bih1, const float* __restrict__ bhh1,
            const float* __restrict__ h_all,   // PASS 0
            const float* __restrict__ aw,      // PASS 1 (softmaxed, [T,G])
            float* __restrict__ logits,        // PASS 0 out
            float* __restrict__ h_att)         // PASS 1 out (atomics)
{
    constexpr int AS = 68;                       // padded LDS row stride (elems)
    __shared__ unsigned short Wl[192 * AS];      // bf16 W_ih1, compact i/g/o rows
    __shared__ float          b1s[192];
    __shared__ float          b0s[192];          // gene-independent layer-0 bias
    __shared__ float          hallT[16 * AS];
    __shared__ unsigned short Atile[8][16 * AS]; // per-wave h0 tile (bf16)
    __shared__ float          coef[8][3 * 64];   // per-wave W_ih0 columns (i,g,o)
    __shared__ float          xrow[8][16];

    const int tid   = threadIdx.x;
    const int lane  = tid & 31;
    const int wv    = tid >> 5;
    const int tbase = blockIdx.x * 16;

    // Stage W_ih1 (only i/g/o gate rows: l in [0,64) U [128,256)) as bf16.
    for (int i = tid; i < 192 * 64; i += 256) {
        int r = i >> 6, c = i & 63;
        int l = (r < 64) ? r : (r + 64);
        Wl[r * AS + c] = f2bf(Wih1[l * 64 + c]);
    }
    if (tid < 192) {
        int l = (tid < 64) ? tid : (tid + 64);
        b1s[tid] = bih1[l] + bhh1[l];
        b0s[tid] = bih0[l] + bhh0[l];
    }
    if (PASS == 0) {
        for (int i = tid; i < 16 * 64; i += 256) {
            int r = i >> 6, c = i & 63;
            hallT[r * AS + c] = h_all[(tbase + r) * 64 + c];
        }
    }
    __syncthreads();

    const int M     = lane & 15;
    const int half  = (lane >> 4) & 1;
    const int khalf = half ? 8 : 0;   // per-lane K offset (ISA 16-bit A layout)
    const int moff  = half ? 8 : 0;   // C-layout row offset

    float*          cf = coef[wv];
    unsigned short* At = Atile[wv];
    float*          xr = xrow[wv];
    const int c0 = lane * 2;          // this lane's two adjacent h0 columns

    v8f hacc[4];
    #pragma unroll
    for (int n = 0; n < 4; ++n)
        #pragma unroll
        for (int r = 0; r < 8; ++r) hacc[n][r] = 0.0f;

    for (int gi = wv; gi < 64; gi += 8) {
        const int g = blockIdx.y * 64 + gi;

        // Per-gene W_ih0 columns -> wave-private LDS via async DMA (ASYNCcnt).
        // Drain pending ds reads of the region we are about to overwrite.
        wait_ds0();
        #pragma unroll
        for (int q = 0; q < 3; ++q) {            // i, g, o gate blocks
            const int lbase = (q == 0) ? 0 : (q == 1 ? 128 : 192);
            async_ld_b32(lds_off(&cf[q * 64 + c0]),
                         &Wih0[(lbase + c0) * G_GENES + g]);
            async_ld_b32(lds_off(&cf[q * 64 + c0 + 1]),
                         &Wih0[(lbase + c0 + 1) * G_GENES + g]);
        }
        if (lane < 16)
            async_ld_b32(lds_off(&xr[lane]), &x[g * T_PTS + tbase + lane]);
        wait_async0();

        // h0 tile [16 x 64] -> bf16 LDS (f gate dead since c0 == 0)
        #pragma unroll 4
        for (int row = 0; row < 16; ++row) {
            float xt = xr[row];
            float ui0 = cf[0 * 64 + c0    ] * xt + b0s[0 * 64 + c0    ];
            float ui1 = cf[0 * 64 + c0 + 1] * xt + b0s[0 * 64 + c0 + 1];
            float ug0 = cf[1 * 64 + c0    ] * xt + b0s[1 * 64 + c0    ];
            float ug1 = cf[1 * 64 + c0 + 1] * xt + b0s[1 * 64 + c0 + 1];
            float uo0 = cf[2 * 64 + c0    ] * xt + b0s[2 * 64 + c0    ];
            float uo1 = cf[2 * 64 + c0 + 1] * xt + b0s[2 * 64 + c0 + 1];
            float h0a = fast_sig(uo0) * fast_tanh(fast_sig(ui0) * fast_tanh(ug0));
            float h0b = fast_sig(uo1) * fast_tanh(fast_sig(ui1) * fast_tanh(ug1));
            *(unsigned*)&At[row * AS + c0] = pack_bf16(h0a, h0b);
        }

        // Build A operands per ISA 16-bit A-matrix 16x32 layout (packed pairs).
        union V16 { v16bf v; unsigned u[8]; };
        V16 A0, A1;
        #pragma unroll
        for (int j = 0; j < 8; ++j) {
            int k = (j < 4) ? (khalf + 2 * j) : (16 + khalf + 2 * (j - 4));
            A0.u[j] = *(const unsigned*)&At[M * AS + k];
            A1.u[j] = *(const unsigned*)&At[M * AS + k + 32];
        }

        float lp[8], awv[8];
        #pragma unroll
        for (int r = 0; r < 8; ++r) lp[r] = 0.0f;
        if (PASS == 1) {
            #pragma unroll
            for (int r = 0; r < 8; ++r)
                awv[r] = aw[(tbase + r + moff) * G_GENES + g];
        }

        #pragma unroll
        for (int n = 0; n < 4; ++n) {       // 16-wide column quarter of H
            v8f acc[3];                     // i, g, o gate accumulators
            #pragma unroll
            for (int q = 0; q < 3; ++q) {
                int bc = q * 64 + n * 16 + M;
                V16 B0, B1;
                #pragma unroll
                for (int j = 0; j < 8; ++j) {
                    int k = (j < 4) ? (khalf + 2 * j) : (16 + khalf + 2 * (j - 4));
                    B0.u[j] = *(const unsigned*)&Wl[bc * AS + k];
                    B1.u[j] = *(const unsigned*)&Wl[bc * AS + k + 32];
                }
                float bias = b1s[q * 64 + n * 16 + M];
                v8f c;
                #pragma unroll
                for (int r = 0; r < 8; ++r) c[r] = bias;
                c = __builtin_amdgcn_wmma_f32_16x16x32_bf16(
                        false, A0.v, false, B0.v, (short)0, c, false, false);
                c = __builtin_amdgcn_wmma_f32_16x16x32_bf16(
                        false, A1.v, false, B1.v, (short)0, c, false, false);
                acc[q] = c;
            }
            #pragma unroll
            for (int r = 0; r < 8; ++r) {
                float hv = fast_sig(acc[2][r]) *
                           fast_tanh(fast_sig(acc[0][r]) * fast_tanh(acc[1][r]));
                if (PASS == 0) {
                    lp[r] += hv * hallT[(r + moff) * AS + n * 16 + M];
                } else {
                    hacc[n][r] += awv[r] * hv;
                }
            }
        }

        if (PASS == 0) {
            // reduce dot over the 16 columns in each lane half, then store
            #pragma unroll
            for (int r = 0; r < 8; ++r) {
                float v = lp[r];
                v = swz_add<0x041F>(v);   // xor 1
                v = swz_add<0x081F>(v);   // xor 2
                v = swz_add<0x101F>(v);   // xor 4
                v = swz_add<0x201F>(v);   // xor 8
                if (M == 0) logits[(tbase + r + moff) * G_GENES + g] = v;
            }
        }
    }

    if (PASS == 1) {
        #pragma unroll
        for (int n = 0; n < 4; ++n)
            #pragma unroll
            for (int r = 0; r < 8; ++r)
                atomicAdd(&h_att[(tbase + r + moff) * H_DIM + n * 16 + M],
                          hacc[n][r]);
    }
}

// ---------------------------------------------------------------------------
// Kernel 3: per-timepoint softmax over genes; writes aw (ws), att^T (out),
// and accumulates imp_score via atomics.
// ---------------------------------------------------------------------------
__global__ __launch_bounds__(256)
void k3_softmax(float* __restrict__ logits,
                float* __restrict__ att_out,   // d_out + T, [G][T]
                float* __restrict__ imp) {     // d_out + T + G*T
    __shared__ float red[256];
    const int t = blockIdx.x, tid = threadIdx.x;
    float* row = logits + (size_t)t * G_GENES;
    float mx = -1e30f;
    for (int g = tid; g < G_GENES; g += 256) mx = fmaxf(mx, row[g]);
    red[tid] = mx; __syncthreads();
    for (int s = 128; s > 0; s >>= 1) {
        if (tid < s) red[tid] = fmaxf(red[tid], red[tid + s]);
        __syncthreads();
    }
    mx = red[0]; __syncthreads();
    float sm = 0.0f;
    for (int g = tid; g < G_GENES; g += 256) sm += __expf(row[g] - mx);
    red[tid] = sm; __syncthreads();
    for (int s = 128; s > 0; s >>= 1) {
        if (tid < s) red[tid] += red[tid + s];
        __syncthreads();
    }
    float rs = __builtin_amdgcn_rcpf(red[0]);
    for (int g = tid; g < G_GENES; g += 256) {
        float a = __expf(row[g] - mx) * rs;
        row[g] = a;                       // ws copy [T,G] for pass 2
        att_out[g * T_PTS + t] = a;       // transposed output [G,T]
        atomicAdd(&imp[g], a * (1.0f / T_PTS));
    }
}

// ---------------------------------------------------------------------------
// Kernel 5: y_pred[t] = h_att[t,:] . W_fc + b_fc
// ---------------------------------------------------------------------------
__global__ void k5_ypred(const float* __restrict__ h_att,
                         const float* __restrict__ Wfc,
                         const float* __restrict__ bfc,
                         float* __restrict__ y) {
    int t = blockIdx.x * blockDim.x + threadIdx.x;
    if (t < T_PTS) {
        float acc = bfc[0];
        #pragma unroll
        for (int h = 0; h < H_DIM; ++h) acc += h_att[t * H_DIM + h] * Wfc[h];
        y[t] = acc;
    }
}

extern "C" void kernel_launch(void* const* d_in, const int* in_sizes, int n_in,
                              void* d_out, int out_size, void* d_ws, size_t ws_size,
                              hipStream_t stream) {
    const float* x    = (const float*)d_in[0];
    const float* Wih0 = (const float*)d_in[1];
    const float* bih0 = (const float*)d_in[2];
    const float* bhh0 = (const float*)d_in[3];
    const float* Wih1 = (const float*)d_in[4];
    const float* bih1 = (const float*)d_in[5];
    const float* bhh1 = (const float*)d_in[6];
    const float* Wfc  = (const float*)d_in[7];
    const float* bfc  = (const float*)d_in[8];

    float* out  = (float*)d_out;
    float* y    = out;                                   // [T]
    float* attT = out + T_PTS;                           // [G][T]
    float* imp  = out + T_PTS + (size_t)G_GENES * T_PTS; // [G]

    float* h_all  = (float*)d_ws;                        // T*H
    float* logits = h_all + T_PTS * H_DIM;               // T*G (-> aw)
    float* h_att  = logits + (size_t)T_PTS * G_GENES;    // T*H

    hipMemsetAsync(h_att, 0, T_PTS * H_DIM * sizeof(float), stream);
    hipMemsetAsync(imp, 0, G_GENES * sizeof(float), stream);

    k1_hall<<<T_PTS, 256, 0, stream>>>(x, Wih0, bih0, bhh0, Wih1, bih1, bhh1, h_all);

    dim3 grid2(T_PTS / 16, G_GENES / 64);
    k_main<0><<<grid2, 256, 0, stream>>>(x, Wih0, bih0, bhh0, Wih1, bih1, bhh1,
                                         h_all, nullptr, logits, nullptr);
    k3_softmax<<<T_PTS, 256, 0, stream>>>(logits, attT, imp);
    k_main<1><<<grid2, 256, 0, stream>>>(x, Wih0, bih0, bhh0, Wih1, bih1, bhh1,
                                         h_all, logits, nullptr, h_att);
    k5_ypred<<<(T_PTS + 255) / 256, 256, 0, stream>>>(h_att, Wfc, bfc, y);
}
